// FusedSelfAttention_17652315587450
// MI455X (gfx1250) — matrix operand order repaired and checked by reading, not compiled, not run on hardware
//
#include <hip/hip_runtime.h>
#include <hip/hip_bf16.h>

typedef __bf16 bf16_t;
typedef __attribute__((ext_vector_type(16))) __bf16 bf16x16;
typedef __attribute__((ext_vector_type(8)))  __bf16 bf16x8;
typedef __attribute__((ext_vector_type(8)))  float  f32x8;

#define S_LEN 4096
#define DIM   2048
#define NH    16
#define HD    128          // head dim
#define QKV_N (3 * DIM)    // 6144

// ---------------------------------------------------------------------------
// WMMA helpers (CDNA5 wave32, 16x16x32 bf16, fp32 accumulate)
// ---------------------------------------------------------------------------
__device__ __forceinline__ f32x8 wmma_bf16(bf16x16 a, bf16x16 b, f32x8 c) {
  // 8 args: (neg_a, A, neg_b, B, c_mod, C, reuse_a, reuse_b)
  return __builtin_amdgcn_wmma_f32_16x16x32_bf16(false, a, false, b, (short)0, c,
                                                 false, false);
}

__device__ __forceinline__ bf16x16 concat8(bf16x8 lo, bf16x8 hi) {
  return __builtin_shufflevector(lo, hi, 0, 1, 2, 3, 4, 5, 6, 7, 8, 9, 10, 11,
                                 12, 13, 14, 15);
}

// A fragment: A row-major [M][lda]. Lane l holds row (row_base + l%16).
// A layout matching load_b: lanes 0-15 hold K = 0..15,
// lanes 16-31 hold K = 16..31 -> two adjacent 16B chunks.
__device__ __forceinline__ bf16x16 load_a(const bf16_t* A, int lda, int row_base,
                                          int k_base, int lane) {
  const int m = row_base + (lane & 15);
  const int h = lane >> 4;
  const bf16_t* p = A + (size_t)m * lda + k_base;
  bf16x8 lo = *(const bf16x8*)(p + 16 * h);
  bf16x8 hi = *(const bf16x8*)(p + 16 * h + 8);
  return concat8(lo, hi);
}

// B fragment: Bt row-major [N][ldb] (i.e. B transposed). Lane l holds column
// (n_base + l%16); same contiguous k order as load_a: lanes 0-15 hold K=0..15, lanes 16-31 hold
// K=16..31 -> one contiguous 32B chunk per lane.
__device__ __forceinline__ bf16x16 load_b(const bf16_t* Bt, int ldb, int n_base,
                                          int k_base, int lane) {
  const int n = n_base + (lane & 15);
  const int h = lane >> 4;
  const bf16_t* p = Bt + (size_t)n * ldb + k_base + 16 * h;
  return *(const bf16x16*)p;
}

// C store: final fp32 output is write-once -> non-temporal (keep L2 for the
// bf16 working set); bf16 intermediates are re-read -> regular temporal.
__device__ __forceinline__ void store_c(float* p, float v) {
  __builtin_nontemporal_store(v, p);
}
__device__ __forceinline__ void store_c(bf16_t* p, float v) { *p = (bf16_t)v; }

// ---------------------------------------------------------------------------
// Precision conversion / layout kernels
// ---------------------------------------------------------------------------
__global__ void cvt_f32_to_bf16(const float* __restrict__ in,
                                bf16_t* __restrict__ out, long n) {
  long i = (long)blockIdx.x * blockDim.x + threadIdx.x;
  if (i < n) out[i] = (bf16_t)in[i];
}

// out[c][r] = in[r][c]; in is [rows_in][cols_in] fp32, out is [cols_in][rows_in] bf16
__global__ void cvt_transpose_bf16(const float* __restrict__ in,
                                   bf16_t* __restrict__ out, int rows_in,
                                   int cols_in) {
  long i = (long)blockIdx.x * blockDim.x + threadIdx.x;
  long total = (long)rows_in * cols_in;
  if (i >= total) return;
  int c = (int)(i / rows_in);
  int r = (int)(i % rows_in);
  out[i] = (bf16_t)in[(size_t)r * cols_in + c];
}

// vt[h*HD+d][s] = qkv[s][2*DIM + h*HD + d]  (V transposed per head)
__global__ void build_vt(const bf16_t* __restrict__ qkv,
                         bf16_t* __restrict__ vt) {
  long i = (long)blockIdx.x * blockDim.x + threadIdx.x;
  if (i >= (long)DIM * S_LEN) return;
  int hd = (int)(i / S_LEN);
  int s = (int)(i % S_LEN);
  vt[i] = qkv[(size_t)s * QKV_N + 2 * DIM + hd];
}

// ---------------------------------------------------------------------------
// Generic bf16 WMMA GEMM: C[M][N] = A[M][K] * Bt[N][K]^T
// One wave (32 threads) per 32x64 output tile: 2 A-frags x 4 B-frags -> 8 WMMA
// per 6 fragment loads per k-step.
// ---------------------------------------------------------------------------
template <typename OUT_T>
__global__ __launch_bounds__(32) void gemm_bf16_wmma(
    const bf16_t* __restrict__ A, const bf16_t* __restrict__ Bt,
    OUT_T* __restrict__ C, int M, int N, int K) {
  const int lane = threadIdx.x;
  const int nb = blockIdx.x * 64;
  const int mb = blockIdx.y * 32;
  f32x8 acc[2][4] = {};
  for (int k0 = 0; k0 < K; k0 += 32) {
    bf16x16 a0 = load_a(A, K, mb, k0, lane);
    bf16x16 a1 = load_a(A, K, mb + 16, k0, lane);
    if (k0 + 32 < K) {
      __builtin_prefetch(A + (size_t)(mb + (lane & 15)) * K + k0 + 32, 0, 0);
      __builtin_prefetch(Bt + (size_t)(nb + (lane & 15)) * K + k0 + 32, 0, 0);
    }
#pragma unroll
    for (int t = 0; t < 4; ++t) {
      bf16x16 b = load_b(Bt, K, nb + 16 * t, k0, lane);
      acc[0][t] = wmma_bf16(a0, b, acc[0][t]);
      acc[1][t] = wmma_bf16(a1, b, acc[1][t]);
    }
  }
  // C/D layout: VGPR r holds row (r + 8*(lane/16)), col = lane%16 within tile
  const int h = lane >> 4, l15 = lane & 15;
#pragma unroll
  for (int mt = 0; mt < 2; ++mt)
#pragma unroll
    for (int t = 0; t < 4; ++t)
#pragma unroll
      for (int r = 0; r < 8; ++r) {
        int row = mb + 16 * mt + r + 8 * h;
        int col = nb + 16 * t + l15;
        store_c(&C[(size_t)row * N + col], acc[mt][t][r]);
      }
}

// ---------------------------------------------------------------------------
// Flash attention step: one 32-key chunk against a 16-row query tile.
// Q in LDS is pre-scaled by log2(e)/sqrt(D), softmax runs in exp2 domain.
// MASKED=false for steps fully inside the causal region (no compares at all).
// ---------------------------------------------------------------------------
template <bool MASKED>
__device__ __forceinline__ void attn_step(int kb, int lane, int l15, int hlf,
                                          int qb, const bf16_t* __restrict__ Kh,
                                          const bf16_t* __restrict__ Vth,
                                          const bf16_t (*q_lds)[136],
                                          bf16_t (*pls)[40], float* m_i,
                                          float* l_i, f32x8* acc) {
  // ---- scores: S(16x32) = Qs(16x128) @ K^T ----
  f32x8 s0 = {}, s1 = {};
#pragma unroll
  for (int kk = 0; kk < 4; ++kk) {
    bf16x8 qlo = *(const bf16x8*)&q_lds[l15][32 * kk + 16 * hlf];
    bf16x8 qhi = *(const bf16x8*)&q_lds[l15][32 * kk + 16 * hlf + 8];
    bf16x16 qa = concat8(qlo, qhi);
    bf16x16 b0 = load_b(Kh, QKV_N, kb, kk * 32, lane);       // keys kb..+15
    bf16x16 b1 = load_b(Kh, QKV_N, kb + 16, kk * 32, lane);  // keys kb+16..+31
    s0 = wmma_bf16(qa, b0, s0);
    s1 = wmma_bf16(qa, b1, s1);
  }
  if (MASKED) {  // causal: key j allowed iff j <= i+1
    const int c0 = kb + l15, c1 = c0 + 16;
#pragma unroll
    for (int r = 0; r < 8; ++r) {
      const int row = qb + r + 8 * hlf;
      if (c0 > row + 1) s0[r] = -3.0e30f;
      if (c1 > row + 1) s1[r] = -3.0e30f;
    }
  }
  // ---- online softmax (log2 domain; rows live in 16-lane halves) ----
#pragma unroll
  for (int r = 0; r < 8; ++r) {
    float mx = fmaxf(s0[r], s1[r]);
    mx = fmaxf(mx, __shfl_xor(mx, 1, 32));
    mx = fmaxf(mx, __shfl_xor(mx, 2, 32));
    mx = fmaxf(mx, __shfl_xor(mx, 4, 32));
    mx = fmaxf(mx, __shfl_xor(mx, 8, 32));
    const float mnew = fmaxf(m_i[r], mx);
    const float corr = exp2f(m_i[r] - mnew);
    const float p0 = exp2f(s0[r] - mnew);
    const float p1 = exp2f(s1[r] - mnew);
    float rs = p0 + p1;
    rs += __shfl_xor(rs, 1, 32);
    rs += __shfl_xor(rs, 2, 32);
    rs += __shfl_xor(rs, 4, 32);
    rs += __shfl_xor(rs, 8, 32);
    l_i[r] = l_i[r] * corr + rs;
    m_i[r] = mnew;
#pragma unroll
    for (int t = 0; t < 8; ++t) acc[t][r] *= corr;
    // stage P (C-layout) into LDS for A-fragment reshape
    pls[r + 8 * hlf][l15] = (bf16_t)p0;
    pls[r + 8 * hlf][16 + l15] = (bf16_t)p1;
  }
  __syncthreads();  // single-wave block: LDS ordering fence
  // ---- O += P(16x32) @ V(32x128) ----
  bf16x8 plo = *(const bf16x8*)&pls[l15][16 * hlf];
  bf16x8 phi = *(const bf16x8*)&pls[l15][16 * hlf + 8];
  bf16x16 pa = concat8(plo, phi);
#pragma unroll
  for (int t = 0; t < 8; ++t) {
    const bf16_t* vp = Vth + (size_t)(t * 16 + l15) * S_LEN + kb + 16 * hlf;
    bf16x16 bv = *(const bf16x16*)vp;
    acc[t] = wmma_bf16(pa, bv, acc[t]);
  }
  __syncthreads();
}

// ---------------------------------------------------------------------------
// Flash attention: one wave per (head, 16-query-row tile).
// ---------------------------------------------------------------------------
__global__ __launch_bounds__(32) void attn_fused(const bf16_t* __restrict__ qkv,
                                                 const bf16_t* __restrict__ vt,
                                                 bf16_t* __restrict__ attn_out) {
  const int lane = threadIdx.x;
  const int l15 = lane & 15, hlf = lane >> 4;
  const int qb = blockIdx.x * 16;  // query row base
  const int hh = blockIdx.y;       // head
  const bf16_t* Qh = qkv + (size_t)hh * HD;
  const bf16_t* Kh = qkv + DIM + (size_t)hh * HD;
  const bf16_t* Vth = vt + (size_t)hh * HD * S_LEN;  // [d][s]
  // log2(e)/sqrt(128): softmax in exp2 domain, scale folded into Q once.
  const float PRESCALE = 0.12747881711546372f;

  // Q tile 16x128 bf16; row stride 136 elems = 272B = 68 dwords -> row r
  // starts at bank 4r, so 16-lane fragment reads hit 64 distinct banks.
  __shared__ __align__(16) bf16_t q_lds[16][136];
  __shared__ __align__(16) bf16_t pls[16][40];  // P tile, 80B row stride

  // stage pre-scaled Q: lane (l15, hlf) copies row l15, cols [64*hlf, +64)
  {
    const bf16_t* src = Qh + (size_t)(qb + l15) * QKV_N + 64 * hlf;
#pragma unroll
    for (int j = 0; j < 64; ++j)
      q_lds[l15][64 * hlf + j] = (bf16_t)((float)src[j] * PRESCALE);
  }
  __syncthreads();

  float m_i[8], l_i[8];
  f32x8 acc[8];
#pragma unroll
  for (int r = 0; r < 8; ++r) { m_i[r] = -3.0e30f; l_i[r] = 0.0f; }
#pragma unroll
  for (int t = 0; t < 8; ++t) acc[t] = (f32x8){};

  int nsteps = (qb + 17 + 31) >> 5;  // keys 0..qb+16 inclusive
  if (nsteps > S_LEN / 32) nsteps = S_LEN / 32;
  // step kb needs no mask iff kb+31 <= qb+1 (all 32 keys allowed for min row)
  int nfull = (qb >= 30) ? (((qb - 30) >> 5) + 1) : 0;
  if (nfull > nsteps) nfull = nsteps;

  int st = 0;
  for (; st < nfull; ++st)
    attn_step<false>(st * 32, lane, l15, hlf, qb, Kh, Vth, q_lds, pls, m_i,
                     l_i, acc);
  for (; st < nsteps; ++st)
    attn_step<true>(st * 32, lane, l15, hlf, qb, Kh, Vth, q_lds, pls, m_i,
                    l_i, acc);

  // ---- normalize and store (bf16, feeds out-proj GEMM) ----
#pragma unroll
  for (int t = 0; t < 8; ++t)
#pragma unroll
    for (int r = 0; r < 8; ++r) {
      const int row = qb + r + 8 * hlf;
      const int col = hh * HD + t * 16 + l15;
      attn_out[(size_t)row * DIM + col] = (bf16_t)(acc[t][r] / l_i[r]);
    }
}

// ---------------------------------------------------------------------------
// Host launcher
// ---------------------------------------------------------------------------
extern "C" void kernel_launch(void* const* d_in, const int* in_sizes, int n_in,
                              void* d_out, int out_size, void* d_ws,
                              size_t ws_size, hipStream_t stream) {
  (void)in_sizes; (void)n_in; (void)out_size; (void)ws_size;
  const float* x = (const float*)d_in[0];      // (4096, 2048)
  const float* w_qkv = (const float*)d_in[1];  // (2048, 6144)
  const float* w_o = (const float*)d_in[2];    // (2048, 2048)
  float* out = (float*)d_out;                  // (4096, 2048) fp32

  char* ws = (char*)d_ws;
  bf16_t* xb = (bf16_t*)ws;      ws += (size_t)S_LEN * DIM * sizeof(bf16_t);
  bf16_t* wqkvT = (bf16_t*)ws;   ws += (size_t)QKV_N * DIM * sizeof(bf16_t);
  bf16_t* woT = (bf16_t*)ws;     ws += (size_t)DIM * DIM * sizeof(bf16_t);
  bf16_t* qkvb = (bf16_t*)ws;    ws += (size_t)S_LEN * QKV_N * sizeof(bf16_t);
  bf16_t* vt = (bf16_t*)ws;      ws += (size_t)DIM * S_LEN * sizeof(bf16_t);
  bf16_t* attn = (bf16_t*)ws;

  // 1) precision / layout prep
  {
    long n = (long)S_LEN * DIM;
    cvt_f32_to_bf16<<<(unsigned)((n + 255) / 256), 256, 0, stream>>>(x, xb, n);
  }
  {
    long n = (long)DIM * QKV_N;
    cvt_transpose_bf16<<<(unsigned)((n + 255) / 256), 256, 0, stream>>>(
        w_qkv, wqkvT, DIM, QKV_N);
  }
  {
    long n = (long)DIM * DIM;
    cvt_transpose_bf16<<<(unsigned)((n + 255) / 256), 256, 0, stream>>>(
        w_o, woT, DIM, DIM);
  }
  // 2) QKV projection: qkv(4096x6144) = x @ w_qkv
  gemm_bf16_wmma<bf16_t><<<dim3(QKV_N / 64, S_LEN / 32), 32, 0, stream>>>(
      xb, wqkvT, qkvb, S_LEN, QKV_N, DIM);
  // 3) V transpose per head for PV B-fragments
  {
    long n = (long)DIM * S_LEN;
    build_vt<<<(unsigned)((n + 255) / 256), 256, 0, stream>>>(qkvb, vt);
  }
  // 4) fused causal flash attention
  attn_fused<<<dim3(S_LEN / 16, NH), 32, 0, stream>>>(qkvb, vt, attn);
  // 5) output projection: out(4096x2048) = attn @ w_o (fp32 result)
  gemm_bf16_wmma<float><<<dim3(DIM / 64, S_LEN / 32), 32, 0, stream>>>(
      attn, woT, out, S_LEN, DIM, DIM);
}
